// PhaseSyncAttentionV6_53850299957949
// MI455X (gfx1250) — matrix operand order repaired and checked by reading, not compiled, not run on hardware
//
#include <hip/hip_runtime.h>

#define DEV __device__ __forceinline__

typedef __attribute__((ext_vector_type(16))) __bf16 v16bf;
typedef __attribute__((ext_vector_type(8)))  float  v8f;

union Frag { v16bf v; unsigned u[8]; };

constexpr int Bn = 4, Ln = 1024, Dn = 768, Hn = 12, Sn = 8, HDn = 64;
constexpr int NTOK = Bn * Ln;                 // 4096
constexpr int WMAT = (Dn / 2) * Dn;           // packed uints per weight matrix

DEV unsigned short f2bf(float f) {
  union { float f; unsigned u; } x; x.f = f;
  unsigned r = x.u + 0x7FFFu + ((x.u >> 16) & 1u);   // round-to-nearest-even
  return (unsigned short)(r >> 16);
}
DEV unsigned pack2(float a, float b) {
  return (unsigned)f2bf(a) | ((unsigned)f2bf(b) << 16);
}
DEV v8f zero8() { v8f z = {0.f,0.f,0.f,0.f,0.f,0.f,0.f,0.f}; return z; }

DEV v8f wmma_bf16(const Frag& a, const Frag& b, v8f c) {
  return __builtin_amdgcn_wmma_f32_16x16x32_bf16(false, a.v, false, b.v,
                                                 (short)0, c, false, false);
}

// A-fragment (16x32 bf16) K offset for VGPR-pair p, lane-half 'half'
DEV int a_koff(int p, int half) {
  return (p < 4) ? (half * 8 + 2 * p) : (16 + half * 8 + 2 * (p - 4));
}

// ------------------------------------------------ one-shot bf16 packing
__global__ void pack_x_kernel(const float* __restrict__ X, unsigned* __restrict__ Xp) {
  int i = blockIdx.x * blockDim.x + threadIdx.x;     // over NTOK*Dn/2
  if (i >= NTOK * Dn / 2) return;
  Xp[i] = pack2(X[2 * i], X[2 * i + 1]);
}

__global__ void pack_w_kernel(const float* __restrict__ Wq, const float* __restrict__ Wk,
                              const float* __restrict__ Wv, const float* __restrict__ Wo,
                              unsigned* __restrict__ Wp) {
  int i = blockIdx.x * blockDim.x + threadIdx.x;     // over 4*WMAT
  if (i >= 4 * WMAT) return;
  int m = i / WMAT, r = i % WMAT;
  int kp = r / Dn, col = r % Dn;
  const float* W = m == 0 ? Wq : m == 1 ? Wk : m == 2 ? Wv : Wo;
  Wp[i] = pack2(W[(size_t)(2 * kp) * Dn + col], W[(size_t)(2 * kp + 1) * Dn + col]);
}

// ---------------------------------------------------------------- features
__global__ void feat_kernel(const float* __restrict__ cosp, const float* __restrict__ sinp,
                            const float* __restrict__ mag,  const float* __restrict__ blog,
                            unsigned short* __restrict__ Pf, float* __restrict__ Mh) {
  int idx = blockIdx.x * blockDim.x + threadIdx.x;   // b*L + l
  if (idx >= Bn * Ln) return;
  int b = idx >> 10, l = idx & (Ln - 1);
  float c[Sn], s[Sn], mg[Sn];
#pragma unroll
  for (int t = 0; t < Sn; ++t) {
    int o = (b * Sn + t) * Ln + l;
    c[t] = cosp[o]; s[t] = sinp[o]; mg[t] = mag[o];
  }
  for (int h = 0; h < Hn; ++h) {
    float bl[Sn], mx = -1e30f;
#pragma unroll
    for (int t = 0; t < Sn; ++t) { bl[t] = blog[h * Sn + t]; mx = fmaxf(mx, bl[t]); }
    float sum = 0.f;
#pragma unroll
    for (int t = 0; t < Sn; ++t) { bl[t] = __expf(bl[t] - mx); sum += bl[t]; }
    float inv = 1.f / sum, mh = 0.f;
    unsigned short* pf = Pf + (size_t)((b * Hn + h) * Ln + l) * 32;
#pragma unroll
    for (int t = 0; t < Sn; ++t) {
      float bw = bl[t] * inv;
      float w  = sqrtf(bw + 1e-8f);
      pf[t]      = f2bf(c[t] * w);
      pf[8 + t]  = f2bf(s[t] * w);
      pf[16 + t] = 0;                 // zero-pad so phase dot fits K=32 WMMA
      pf[24 + t] = 0;
      mh += mg[t] * bw;
    }
    Mh[(b * Hn + h) * Ln + l] = mh;
  }
}

// ---------------------------------------------------------------- QKV GEMM (bf16 in, bf16 out)
__global__ __launch_bounds__(128)
void qkv_kernel(const unsigned* __restrict__ Xp, const unsigned* __restrict__ Wp,
                const float* __restrict__ bq, const float* __restrict__ bk,
                const float* __restrict__ bv,
                unsigned short* __restrict__ Qb, unsigned short* __restrict__ Kb,
                unsigned short* __restrict__ Vb) {
  int lane = threadIdx.x & 31, wave = threadIdx.x >> 5;
  int half = lane >> 4, n = lane & 15;
  int id = blockIdx.x;                       // [which][rowT][cb]
  int which = id / (256 * 3);
  int rem = id % (256 * 3);
  int rowT = rem / 3, cb = rem % 3;
  int row0 = rowT * 16;
  int col0 = cb * 256 + wave * 64;
  const unsigned* W = Wp + (size_t)which * WMAT;
  const float* bias = which == 0 ? bq : which == 1 ? bk : bv;
  unsigned short* dst = which == 0 ? Qb : which == 1 ? Kb : Vb;

  v8f acc[4] = { zero8(), zero8(), zero8(), zero8() };
  int rowA = row0 + n;
  const unsigned* xrow = Xp + (size_t)rowA * (Dn / 2);
  for (int kc = 0; kc < Dn / 32; ++kc) {
    Frag a;
#pragma unroll
    for (int p = 0; p < 8; ++p)
      a.u[p] = xrow[(kc * 32 + a_koff(p, half)) >> 1];
#pragma unroll
    for (int nt = 0; nt < 4; ++nt) {
      Frag bB;
      int colc = col0 + nt * 16 + n;
#pragma unroll
      for (int p = 0; p < 8; ++p) {
        int kp = kc * 16 + (a_koff(p, half) >> 1);     // packed (2kp, 2kp+1) pair
        bB.u[p] = W[(size_t)kp * Dn + colc];
      }
      acc[nt] = wmma_bf16(a, bB, acc[nt]);
    }
  }
#pragma unroll
  for (int nt = 0; nt < 4; ++nt) {
    int col = col0 + nt * 16 + n;
    int hh = col >> 6, hd = col & 63;
    float bs = bias[col];
#pragma unroll
    for (int e = 0; e < 8; ++e) {
      int row = row0 + e + 8 * half;
      int bi = row >> 10, li = row & (Ln - 1);
      dst[(size_t)(((bi * Hn + hh) * Ln + li)) * HDn + hd] = f2bf(acc[nt][e] + bs);
    }
  }
}

// ---------------------------------------------------------------- fused attention
__global__ __launch_bounds__(128)
void attn_kernel(const unsigned short* __restrict__ Qb, const unsigned short* __restrict__ Kb,
                 const unsigned short* __restrict__ Vb, const unsigned short* __restrict__ Pf,
                 const float* __restrict__ Mh, const float* __restrict__ mask,
                 const float* __restrict__ pbias, const float* __restrict__ gscale,
                 unsigned short* __restrict__ Cb) {
  __shared__ unsigned short Ks[32 * 64];       // [token][hd]
  __shared__ unsigned short Vs[64 * 32];       // transposed [hd][token]
  __shared__ unsigned short Ps[32 * 32];       // [token][feat]
  __shared__ unsigned short Wb[4][16 * 32];    // per-wave softmax weights [M][key]
  __shared__ float Ms[32], Mk[32];

  int tid = threadIdx.x, lane = tid & 31, wave = tid >> 5;
  int half = lane >> 4, n = lane & 15;
  int blk = blockIdx.x;
  int qblk = blk & 15;            // L/64 tiles
  int bh = blk >> 4;              // b*H + h
  int h = bh % Hn, b = bh / Hn;
  int q0 = qblk * 64 + wave * 16;

  const unsigned* Qu = (const unsigned*)Qb;
  const unsigned* Pu = (const unsigned*)Pf;

  Frag qA[2], pA;
#pragma unroll
  for (int c = 0; c < 2; ++c)
#pragma unroll
    for (int p = 0; p < 8; ++p) {
      int ko = a_koff(p, half);
      qA[c].u[p] = Qu[((size_t)(bh * Ln + q0 + n) * HDn + c * 32 + ko) >> 1];
    }
#pragma unroll
  for (int p = 0; p < 8; ++p) {
    int ko = a_koff(p, half);
    pA.u[p] = Pu[((size_t)(bh * Ln + q0 + n) * 32 + ko) >> 1];
  }
  float magq[8];
#pragma unroll
  for (int e = 0; e < 8; ++e) magq[e] = Mh[bh * Ln + q0 + e + 8 * half];
  float pscale = __expf(pbias[h]);
  float g = gscale[0];

  v8f acc[4] = { zero8(), zero8(), zero8(), zero8() };
  float m[8], l[8];
#pragma unroll
  for (int e = 0; e < 8; ++e) { m[e] = -1e30f; l[e] = 0.f; }

  for (int j0 = 0; j0 < Ln; j0 += 32) {
    __syncthreads();
    { // cooperative staging of K/V/features for 32 keys
      const unsigned* Ku = (const unsigned*)Kb + (size_t)(bh * Ln + j0) * 32;
      const unsigned* Vu = (const unsigned*)Vb + (size_t)(bh * Ln + j0) * 32;
      const unsigned* Pg = (const unsigned*)Pf + (size_t)(bh * Ln + j0) * 16;
      unsigned* KsU = (unsigned*)Ks;
      unsigned* PsU = (unsigned*)Ps;
      for (int i = tid; i < 1024; i += 128) KsU[i] = Ku[i];
      for (int i = tid; i < 512;  i += 128) PsU[i] = Pg[i];
      for (int i = tid; i < 1024; i += 128) {
        unsigned pv = Vu[i];
        int tok = i >> 5, hd = (i & 31) * 2;
        Vs[hd * 32 + tok]       = (unsigned short)(pv & 0xFFFFu);
        Vs[(hd + 1) * 32 + tok] = (unsigned short)(pv >> 16);
      }
      if (tid < 32) {
        Ms[tid] = Mh[bh * Ln + j0 + tid];
        Mk[tid] = mask[b * Ln + j0 + tid];
      }
    }
    __syncthreads();

    const unsigned* KsU = (const unsigned*)Ks;
    const unsigned* PsU = (const unsigned*)Ps;
    for (int t = 0; t < 2; ++t) {
      Frag kB0, kB1, pB;
      int tok = t * 16 + n;
#pragma unroll
      for (int p = 0; p < 8; ++p) {
        int kk = a_koff(p, half);
        kB0.u[p] = KsU[(tok * 64 + kk) >> 1];
        kB1.u[p] = KsU[(tok * 64 + 32 + kk) >> 1];
        pB.u[p]  = PsU[(tok * 32 + kk) >> 1];
      }
      v8f s = wmma_bf16(qA[0], kB0, zero8());
      s = wmma_bf16(qA[1], kB1, s);
      v8f ph = wmma_bf16(pA, pB, zero8());
      float mj = Ms[t * 16 + n];
      float mk = Mk[t * 16 + n];
#pragma unroll
      for (int e = 0; e < 8; ++e) {
        float mi = magq[e];
        float hm = 2.f * mi * mj / (mi + mj + 1e-8f);
        float pv = ph[e] * hm * pscale;
        float sc = s[e] * 0.125f + pv + mk;          // 1/sqrt(64)
        float gt = 1.f / (1.f + __expf(-g * pv));    // sigmoid gate
        float rm = sc;                               // row max across 16 lanes
        rm = fmaxf(rm, __shfl_xor(rm, 1));
        rm = fmaxf(rm, __shfl_xor(rm, 2));
        rm = fmaxf(rm, __shfl_xor(rm, 4));
        rm = fmaxf(rm, __shfl_xor(rm, 8));
        float mnew = fmaxf(m[e], rm);
        float esc = __expf(m[e] - mnew);
        float w = __expf(sc - mnew) * gt;
        float rs = w;
        rs += __shfl_xor(rs, 1);
        rs += __shfl_xor(rs, 2);
        rs += __shfl_xor(rs, 4);
        rs += __shfl_xor(rs, 8);
        l[e] = l[e] * esc + rs;
        m[e] = mnew;
        acc[0][e] *= esc; acc[1][e] *= esc; acc[2][e] *= esc; acc[3][e] *= esc;
        Wb[wave][(e + 8 * half) * 32 + tok] = f2bf(w);
      }
    }
    __syncthreads();

    // P·V over 32 keys: re-layout weights C-tile -> A-fragment through LDS
    Frag aW;
    const unsigned* WbU = (const unsigned*)Wb[wave];
#pragma unroll
    for (int p = 0; p < 8; ++p) {
      int ko = a_koff(p, half);
      aW.u[p] = WbU[(n * 32 + ko) >> 1];
    }
    const unsigned* VsU = (const unsigned*)Vs;
#pragma unroll
    for (int nt = 0; nt < 4; ++nt) {
      Frag bV;
#pragma unroll
      for (int p = 0; p < 8; ++p) {
        int key = a_koff(p, half);
        bV.u[p] = VsU[((nt * 16 + n) * 32 + key) >> 1];
      }
      acc[nt] = wmma_bf16(aW, bV, acc[nt]);
    }
  }

#pragma unroll
  for (int e = 0; e < 8; ++e) {
    float inv = 1.f / (l[e] + 1e-9f);
    int row = q0 + e + 8 * half;
#pragma unroll
    for (int nt = 0; nt < 4; ++nt)
      Cb[(size_t)(bh * Ln + row) * HDn + nt * 16 + n] = f2bf(acc[nt][e] * inv);
  }
}

// ------------------------------------------- output projection (+bias+residual)
__global__ __launch_bounds__(128)
void proj_kernel(const unsigned short* __restrict__ Cb, const unsigned* __restrict__ Wp,
                 const float* __restrict__ bo, const float* __restrict__ X,
                 float* __restrict__ Tp) {
  int lane = threadIdx.x & 31, wave = threadIdx.x >> 5;
  int half = lane >> 4, n = lane & 15;
  int id = blockIdx.x;
  int rowT = id / 3, cb = id % 3;
  int row0 = rowT * 16, col0 = cb * 256 + wave * 64;
  const unsigned* Cu = (const unsigned*)Cb;
  const unsigned* W = Wp + (size_t)3 * WMAT;          // packed Wo
  v8f acc[4] = { zero8(), zero8(), zero8(), zero8() };
  int rowA = row0 + n;
  int bi = rowA >> 10, li = rowA & (Ln - 1);
  for (int kc = 0; kc < Dn / 32; ++kc) {
    Frag a;
#pragma unroll
    for (int p = 0; p < 8; ++p) {
      int k = kc * 32 + a_koff(p, half);
      int hh = k >> 6, hd = k & 63;
      a.u[p] = Cu[((size_t)(((bi * Hn + hh) * Ln + li)) * HDn + hd) >> 1];
    }
#pragma unroll
    for (int nt = 0; nt < 4; ++nt) {
      Frag bB;
      int colc = col0 + nt * 16 + n;
#pragma unroll
      for (int p = 0; p < 8; ++p) {
        int kp = kc * 16 + (a_koff(p, half) >> 1);
        bB.u[p] = W[(size_t)kp * Dn + colc];
      }
      acc[nt] = wmma_bf16(a, bB, acc[nt]);
    }
  }
#pragma unroll
  for (int nt = 0; nt < 4; ++nt) {
    int col = col0 + nt * 16 + n;
    float bs = bo[col];
#pragma unroll
    for (int e = 0; e < 8; ++e) {
      int row = row0 + e + 8 * half;
      Tp[(size_t)row * Dn + col] = acc[nt][e] + bs + X[(size_t)row * Dn + col];
    }
  }
}

// ---------------------------------------------------------------- LayerNorm
__global__ void ln_kernel(const float* __restrict__ Tp, const float* __restrict__ gamma,
                          const float* __restrict__ beta, float* __restrict__ out) {
  __shared__ float s1[256], s2[256];
  int row = blockIdx.x, tid = threadIdx.x;
  const float* x = Tp + (size_t)row * Dn;
  float a = 0.f, bq = 0.f;
  for (int i = tid; i < Dn; i += 256) { float v = x[i]; a += v; bq += v * v; }
  s1[tid] = a; s2[tid] = bq;
  __syncthreads();
  for (int st = 128; st > 0; st >>= 1) {
    if (tid < st) { s1[tid] += s1[tid + st]; s2[tid] += s2[tid + st]; }
    __syncthreads();
  }
  float mu = s1[0] * (1.f / Dn);
  float var = s2[0] * (1.f / Dn) - mu * mu;
  float rstd = rsqrtf(var + 1e-12f);
  float* o = out + (size_t)row * Dn;
  for (int i = tid; i < Dn; i += 256) o[i] = (x[i] - mu) * rstd * gamma[i] + beta[i];
}

// ---------------------------------------------------------------- launch
extern "C" void kernel_launch(void* const* d_in, const int* in_sizes, int n_in,
                              void* d_out, int out_size, void* d_ws, size_t ws_size,
                              hipStream_t stream) {
  const float* X     = (const float*)d_in[0];
  const float* mask  = (const float*)d_in[1];
  const float* cosp  = (const float*)d_in[2];
  const float* sinp  = (const float*)d_in[3];
  const float* mag   = (const float*)d_in[4];
  const float* Wq    = (const float*)d_in[5];
  const float* bq    = (const float*)d_in[6];
  const float* Wk    = (const float*)d_in[7];
  const float* bk    = (const float*)d_in[8];
  const float* Wv    = (const float*)d_in[9];
  const float* bvv   = (const float*)d_in[10];
  const float* Wo    = (const float*)d_in[11];
  const float* bo    = (const float*)d_in[12];
  const float* gamma = (const float*)d_in[13];
  const float* beta  = (const float*)d_in[14];
  const float* blog  = (const float*)d_in[15];
  const float* pbias = (const float*)d_in[16];
  const float* gsc   = (const float*)d_in[17];

  char* ws = (char*)d_ws;
  size_t off = 0;
  auto alloc = [&](size_t bytes) -> char* {
    char* p = ws + off;
    off += (bytes + 255) & ~(size_t)255;
    return p;
  };
  unsigned short* Qb = (unsigned short*)alloc((size_t)NTOK * Dn * 2);
  unsigned short* Kb = (unsigned short*)alloc((size_t)NTOK * Dn * 2);
  unsigned short* Vb = (unsigned short*)alloc((size_t)NTOK * Dn * 2);
  unsigned short* Cb = (unsigned short*)alloc((size_t)NTOK * Dn * 2);
  unsigned short* Pf = (unsigned short*)alloc((size_t)Bn * Hn * Ln * 32 * 2);
  float*          Mh = (float*)alloc((size_t)Bn * Hn * Ln * 4);
  float*          Tp = (float*)alloc((size_t)NTOK * Dn * 4);
  unsigned*       Xp = (unsigned*)alloc((size_t)NTOK * (Dn / 2) * 4);
  unsigned*       Wp = (unsigned*)alloc((size_t)4 * WMAT * 4);
  (void)ws_size; (void)in_sizes; (void)n_in; (void)out_size;

  pack_x_kernel<<<(NTOK * Dn / 2 + 255) / 256, 256, 0, stream>>>(X, Xp);
  pack_w_kernel<<<(4 * WMAT + 255) / 256, 256, 0, stream>>>(Wq, Wk, Wv, Wo, Wp);
  feat_kernel<<<(Bn * Ln) / 256, 256, 0, stream>>>(cosp, sinp, mag, blog, Pf, Mh);
  qkv_kernel<<<3 * 256 * 3, 128, 0, stream>>>(Xp, Wp, bq, bk, bvv, Qb, Kb, Vb);
  attn_kernel<<<Bn * Hn * (Ln / 64), 128, 0, stream>>>(Qb, Kb, Vb, Pf, Mh, mask, pbias, gsc, Cb);
  proj_kernel<<<256 * 3, 128, 0, stream>>>(Cb, Wp, bo, X, Tp);
  ln_kernel<<<NTOK, 256, 0, stream>>>(Tp, gamma, beta, (float*)d_out);
}
